// HybridAttention_83442624626991
// MI455X (gfx1250) — matrix operand order repaired and checked by reading, not compiled, not run on hardware
//
#include <hip/hip_runtime.h>

// ---------------------------------------------------------------------------
// HybridAttention (global MHA + banded local MHA) for MI455X / gfx1250.
// All GEMMs run on v_wmma_f32_16x16x32_f16 (f16 in, f32 accumulate).
// Flash-style single-pass attention; local branch computes only the band.
// K/V tiles staged LDS-direct with global_load_async_to_lds_b128 (ASYNCcnt).
// Workspace layout (f16): Qg Kg Vg Ql Kl Vl Og Ol, each B*H*S*K = 4,194,304
// elements -> 8 MiB each, 64 MiB total.
// ---------------------------------------------------------------------------

typedef __attribute__((ext_vector_type(16))) _Float16 v16h;
typedef __attribute__((ext_vector_type(8)))  float    v8f;

static constexpr int Bv = 2, Sv = 2048, Dv = 128, Hv = 8, Kv = 128, Wv = 256;
static constexpr float SCALE = 0.08838834764831845f;  // 1/sqrt(128)
static constexpr float NEG_BIG = -1e30f;

// ---- CDNA5 WMMA fragment coordinate maps (CDNA5 ISA §7.12.2) ----
// 16-bit A 16x32 (MxK), contiguous k order used here: M = lane&15; lanes 0..15 hold K=0..15 (slot
// order), lanes 16..31 hold K=16..31 (matches b_coords).
__device__ __forceinline__ void a_coords(int lane, int s, int& m, int& k) {
  m = lane & 15;
  k = (lane & 16) + s;
}
// 16-bit B 32x16 (KxN), same contiguous k order: lanes 0..15 hold K=0..15 (slot order),
// lanes 16..31 hold K=16..31; N = lane&15.
__device__ __forceinline__ void b_coords(int lane, int s, int& k, int& n) {
  k = (lane & 16) + s;
  n = lane & 15;
}
// f32 C/D 16x16: VGPR r -> M = r (lanes 0..15) / r+8 (lanes 16..31), N=lane&15.
__device__ __forceinline__ void cd_coords(int lane, int r, int& m, int& n) {
  m = r + ((lane & 16) ? 8 : 0);
  n = lane & 15;
}

__device__ __forceinline__ v16h load_a_f32(const float* __restrict__ p, int ld) {
  v16h a; const int lane = threadIdx.x & 31;
#pragma unroll
  for (int s = 0; s < 16; ++s) { int m, k; a_coords(lane, s, m, k); a[s] = (_Float16)p[m * ld + k]; }
  return a;
}
__device__ __forceinline__ v16h load_a_f16(const _Float16* __restrict__ p, int ld) {
  v16h a; const int lane = threadIdx.x & 31;
#pragma unroll
  for (int s = 0; s < 16; ++s) { int m, k; a_coords(lane, s, m, k); a[s] = p[m * ld + k]; }
  return a;
}
// B(k,n) = p[k*ld + n]
__device__ __forceinline__ v16h load_b_f32(const float* __restrict__ p, int ld) {
  v16h bf; const int lane = threadIdx.x & 31;
#pragma unroll
  for (int s = 0; s < 16; ++s) { int k, n; b_coords(lane, s, k, n); bf[s] = (_Float16)p[k * ld + n]; }
  return bf;
}
__device__ __forceinline__ v16h load_b_f16(const _Float16* __restrict__ p, int ld) {
  v16h bf; const int lane = threadIdx.x & 31;
#pragma unroll
  for (int s = 0; s < 16; ++s) { int k, n; b_coords(lane, s, k, n); bf[s] = p[k * ld + n]; }
  return bf;
}
// Transposed gather: B(k,n) = p[n*ld + k]  (used for K^T from row-major K)
__device__ __forceinline__ v16h load_bt_f16(const _Float16* __restrict__ p, int ld) {
  v16h bf; const int lane = threadIdx.x & 31;
#pragma unroll
  for (int s = 0; s < 16; ++s) { int k, n; b_coords(lane, s, k, n); bf[s] = p[n * ld + k]; }
  return bf;
}

__device__ __forceinline__ v8f wmma16(v16h a, v16h b, v8f c) {
  return __builtin_amdgcn_wmma_f32_16x16x32_f16(false, a, false, b, (short)0, c,
                                                false, false);
}

// Async global->LDS copy of 32 bytes (2x b128) per thread. lds_off is the
// byte offset within the workgroup LDS allocation (= low 32 bits of the
// generic shared pointer).
// Tracked by ASYNCcnt; hardware adds INST_OFFSET to both LDS and global addr.
__device__ __forceinline__ void async_copy32(unsigned lds_off, const void* gaddr) {
  asm volatile(
      "global_load_async_to_lds_b128 %0, %1, off\n\t"
      "global_load_async_to_lds_b128 %0, %1, off offset:16"
      :: "v"(lds_off), "v"(gaddr) : "memory");
}
__device__ __forceinline__ void wait_async0() {
  asm volatile("s_wait_asynccnt 0" ::: "memory");
}

// ---------------------------------------------------------------------------
// QKV projection: P[bh, s, k] = sum_d X[b,s,d] * W[d,h,k] + bias[h,k]  (f16 out)
// One wave computes a 16(s) x 128(k) strip.
// ---------------------------------------------------------------------------
__global__ __launch_bounds__(32) void ha_proj_kernel(
    const float* __restrict__ X, const float* __restrict__ Wm,
    const float* __restrict__ bias, _Float16* __restrict__ P) {
  const int bh = blockIdx.y;
  const int b = bh / Hv, h = bh % Hv;
  const int s0 = blockIdx.x * 16;
  const int lane = threadIdx.x & 31;

  const float* Xb = X + ((size_t)b * Sv + s0) * Dv;
  v16h af[4];
#pragma unroll
  for (int kc = 0; kc < 4; ++kc) af[kc] = load_a_f32(Xb + kc * 32, Dv);

#pragma unroll
  for (int nt = 0; nt < 8; ++nt) {
    v8f c = {};
#pragma unroll
    for (int kc = 0; kc < 4; ++kc) {
      v16h bf;
#pragma unroll
      for (int s = 0; s < 16; ++s) {
        int k, n; b_coords(lane, s, k, n);
        bf[s] = (_Float16)Wm[(size_t)(kc * 32 + k) * (Hv * Kv) + h * Kv + nt * 16 + n];
      }
      c = wmma16(af[kc], bf, c);
    }
    const float bvv = bias[h * Kv + nt * 16 + (lane & 15)];
#pragma unroll
    for (int r = 0; r < 8; ++r) {
      int m, n; cd_coords(lane, r, m, n);
      P[((size_t)bh * Sv + s0 + m) * Kv + nt * 16 + n] = (_Float16)(c[r] + bvv);
    }
  }
}

// ---------------------------------------------------------------------------
// Flash attention: 8 waves/block, each owns a 16-row query tile of a 128-row
// block; K/V staged in LDS in chunks of 32 keys via async-to-LDS loads;
// online softmax in registers. local!=0 masks to |q-s| <= W/2.
// ---------------------------------------------------------------------------
__global__ __launch_bounds__(256) void ha_attn_kernel(
    const _Float16* __restrict__ Q, const _Float16* __restrict__ Kh,
    const _Float16* __restrict__ Vh, _Float16* __restrict__ O, int local) {
  const int bh = blockIdx.y;
  const int b = bh / Hv, h = bh % Hv;
  const int q0 = blockIdx.x * 128;
  const int wave = threadIdx.x >> 5;
  const int lane = threadIdx.x & 31;
  const int qt = q0 + wave * 16;

  __shared__ __align__(16) _Float16 sK[32 * Kv];
  __shared__ __align__(16) _Float16 sV[32 * Kv];
  __shared__ __align__(16) _Float16 sP[8][16 * 32];

  const _Float16* Qb = Q + ((size_t)bh * Sv + qt) * Kv;
  v16h qa[4];
#pragma unroll
  for (int kc = 0; kc < 4; ++kc) qa[kc] = load_a_f16(Qb + kc * 32, Kv);

  v8f o[8];
#pragma unroll
  for (int j = 0; j < 8; ++j) o[j] = {};
  float mrow[8], lrow[8];
#pragma unroll
  for (int r = 0; r < 8; ++r) { mrow[r] = NEG_BIG; lrow[r] = 0.f; }

  int sbeg = 0, send = Sv;
  if (local) {
    sbeg = q0 - Wv / 2; if (sbeg < 0) sbeg = 0;
    send = q0 + 128 + Wv / 2; if (send > Sv) send = Sv;
  }

  const unsigned kof = (unsigned)(uintptr_t)(&sK[0]) + (unsigned)threadIdx.x * 32u;
  const unsigned vof = (unsigned)(uintptr_t)(&sV[0]) + (unsigned)threadIdx.x * 32u;

  for (int s0 = sbeg; s0 < send; s0 += 32) {
    {  // async stage: 32 keys x 128 feats of K and V (8 KiB each), LDS-direct
      const char* kg = (const char*)(Kh + ((size_t)bh * Sv + s0) * Kv) + threadIdx.x * 32;
      const char* vg = (const char*)(Vh + ((size_t)bh * Sv + s0) * Kv) + threadIdx.x * 32;
      async_copy32(kof, kg);
      async_copy32(vof, vg);
      if (s0 + 32 < send)  // gfx1250 global_prefetch_b8 for next chunk
        __builtin_prefetch((const void*)(Kh + ((size_t)bh * Sv + s0 + 32) * Kv), 0, 3);
      wait_async0();       // this wave's async loads landed in LDS
    }
    __syncthreads();       // all waves' chunks visible

    const bool active =
        !local || ((s0 + 31 >= qt - Wv / 2) && (s0 <= qt + 15 + Wv / 2));
    if (active) {
      // ---- scores: 16(q) x 32(s) via two 16x16 accumulators ----
      v8f sc0 = {}, sc1 = {};
#pragma unroll
      for (int kc = 0; kc < 4; ++kc) {
        sc0 = wmma16(qa[kc], load_bt_f16(sK + 0 * 16 * Kv + kc * 32, Kv), sc0);
        sc1 = wmma16(qa[kc], load_bt_f16(sK + 1 * 16 * Kv + kc * 32, Kv), sc1);
      }
      const int col = lane & 15;
      const int row0 = qt + ((lane & 16) ? 8 : 0);
#pragma unroll
      for (int r = 0; r < 8; ++r) {
        float a0 = sc0[r] * SCALE, a1 = sc1[r] * SCALE;
        if (local) {
          int d0 = (row0 + r) - (s0 + col);      if (d0 < 0) d0 = -d0;
          int d1 = (row0 + r) - (s0 + 16 + col); if (d1 < 0) d1 = -d1;
          if (d0 > Wv / 2) a0 = NEG_BIG;
          if (d1 > Wv / 2) a1 = NEG_BIG;
        }
        sc0[r] = a0; sc1[r] = a1;
      }
      // ---- online softmax (row = C-layout component; 16-lane reductions) ----
#pragma unroll
      for (int r = 0; r < 8; ++r) {
        float t = fmaxf(sc0[r], sc1[r]);
        t = fmaxf(t, __shfl_xor(t, 1, 32));
        t = fmaxf(t, __shfl_xor(t, 2, 32));
        t = fmaxf(t, __shfl_xor(t, 4, 32));
        t = fmaxf(t, __shfl_xor(t, 8, 32));
        const float mn = fmaxf(mrow[r], t);
        const float alpha = __expf(mrow[r] - mn);
        const float p0 = __expf(sc0[r] - mn);
        const float p1 = __expf(sc1[r] - mn);
        sc0[r] = p0; sc1[r] = p1;
        float rs = p0 + p1;
        rs += __shfl_xor(rs, 1, 32);
        rs += __shfl_xor(rs, 2, 32);
        rs += __shfl_xor(rs, 4, 32);
        rs += __shfl_xor(rs, 8, 32);
        lrow[r] = lrow[r] * alpha + rs;
        mrow[r] = mn;
#pragma unroll
        for (int j = 0; j < 8; ++j) o[j][r] *= alpha;
      }
      // ---- C-layout -> A-layout via per-wave LDS bounce, then PV WMMA ----
      _Float16* myP = &sP[wave][0];
#pragma unroll
      for (int r = 0; r < 8; ++r) {
        const int m = r + ((lane & 16) ? 8 : 0);
        myP[m * 32 + col] = (_Float16)sc0[r];
        myP[m * 32 + 16 + col] = (_Float16)sc1[r];
      }
      const v16h pa = load_a_f16(myP, 32);
#pragma unroll
      for (int j = 0; j < 8; ++j)
        o[j] = wmma16(pa, load_b_f16(sV + j * 16, Kv), o[j]);
    }
    __syncthreads();
  }

  // ---- normalize and store O[b, q, h*K + k] as f16 ----
  _Float16* Ob = O + (size_t)b * Sv * (Hv * Kv) + h * Kv;
#pragma unroll
  for (int j = 0; j < 8; ++j)
#pragma unroll
    for (int r = 0; r < 8; ++r) {
      int m, n; cd_coords(lane, r, m, n);
      Ob[(size_t)(qt + m) * (Hv * Kv) + j * 16 + n] = (_Float16)(o[j][r] / lrow[r]);
    }
}

// ---------------------------------------------------------------------------
// Fused output projection of BOTH branches + biases:
// out[b,s,d] = sum_c Og[b,s,c]*Wg[c,d] + sum_c Ol[b,s,c]*Wl[c,d] + bg[d]+bl[d]
// ---------------------------------------------------------------------------
__global__ __launch_bounds__(32) void ha_oproj_kernel(
    const _Float16* __restrict__ Og, const _Float16* __restrict__ Ol,
    const float* __restrict__ Wg, const float* __restrict__ Wl,
    const float* __restrict__ bg, const float* __restrict__ bl,
    float* __restrict__ out) {
  const int d0 = blockIdx.x * 16;
  const int s0 = blockIdx.y * 16;
  const int b = blockIdx.z;
  const int lane = threadIdx.x & 31;

  v8f c = {};
  const _Float16* Os[2] = {Og, Ol};
  const float* Ws[2] = {Wg, Wl};
#pragma unroll
  for (int br = 0; br < 2; ++br) {
    const _Float16* Ob = Os[br] + ((size_t)b * Sv + s0) * (Hv * Kv);
    for (int kc = 0; kc < 32; ++kc) {  // 1024 / 32
      const v16h a = load_a_f16(Ob + kc * 32, Hv * Kv);
      const v16h bf = load_b_f32(Ws[br] + (size_t)kc * 32 * Dv + d0, Dv);
      c = wmma16(a, bf, c);
    }
  }
  const float bvv = bg[d0 + (lane & 15)] + bl[d0 + (lane & 15)];
#pragma unroll
  for (int r = 0; r < 8; ++r) {
    int m, n; cd_coords(lane, r, m, n);
    out[((size_t)b * Sv + s0 + m) * Dv + d0 + n] = c[r] + bvv;
  }
}

// ---------------------------------------------------------------------------
extern "C" void kernel_launch(void* const* d_in, const int* in_sizes, int n_in,
                              void* d_out, int out_size, void* d_ws, size_t ws_size,
                              hipStream_t stream) {
  (void)in_sizes; (void)n_in; (void)out_size; (void)ws_size;

  const float* query = (const float*)d_in[0];
  const float* key   = (const float*)d_in[1];
  const float* value = (const float*)d_in[2];
  const float* gq_w = (const float*)d_in[3];  const float* gq_b = (const float*)d_in[4];
  const float* gk_w = (const float*)d_in[5];  const float* gk_b = (const float*)d_in[6];
  const float* gv_w = (const float*)d_in[7];  const float* gv_b = (const float*)d_in[8];
  const float* go_w = (const float*)d_in[9];  const float* go_b = (const float*)d_in[10];
  const float* lq_w = (const float*)d_in[11]; const float* lq_b = (const float*)d_in[12];
  const float* lk_w = (const float*)d_in[13]; const float* lk_b = (const float*)d_in[14];
  const float* lv_w = (const float*)d_in[15]; const float* lv_b = (const float*)d_in[16];
  const float* lo_w = (const float*)d_in[17]; const float* lo_b = (const float*)d_in[18];

  const size_t N = (size_t)Bv * Hv * Sv * Kv;  // 4,194,304 f16 per tensor
  _Float16* ws = (_Float16*)d_ws;
  _Float16* Qg = ws + 0 * N; _Float16* Kg = ws + 1 * N; _Float16* Vg = ws + 2 * N;
  _Float16* Ql = ws + 3 * N; _Float16* Kl = ws + 4 * N; _Float16* Vl = ws + 5 * N;
  _Float16* Og = ws + 6 * N; _Float16* Ol = ws + 7 * N;

  const dim3 pgrid(Sv / 16, Bv * Hv), pblk(32);
  ha_proj_kernel<<<pgrid, pblk, 0, stream>>>(query, gq_w, gq_b, Qg);
  ha_proj_kernel<<<pgrid, pblk, 0, stream>>>(key,   gk_w, gk_b, Kg);
  ha_proj_kernel<<<pgrid, pblk, 0, stream>>>(value, gv_w, gv_b, Vg);
  ha_proj_kernel<<<pgrid, pblk, 0, stream>>>(query, lq_w, lq_b, Ql);
  ha_proj_kernel<<<pgrid, pblk, 0, stream>>>(key,   lk_w, lk_b, Kl);
  ha_proj_kernel<<<pgrid, pblk, 0, stream>>>(value, lv_w, lv_b, Vl);

  const dim3 agrid(Sv / 128, Bv * Hv), ablk(256);
  ha_attn_kernel<<<agrid, ablk, 0, stream>>>(Qg, Kg, Vg, Og, 0);
  ha_attn_kernel<<<agrid, ablk, 0, stream>>>(Ql, Kl, Vl, Ol, 1);

  const dim3 ogrid(Dv / 16, Sv / 16, Bv), oblk(32);
  ha_oproj_kernel<<<ogrid, oblk, 0, stream>>>(Og, Ol, go_w, lo_w, go_b, lo_b,
                                              (float*)d_out);
}
